// TrajPreAttnAvgLongUser_16063177687446
// MI455X (gfx1250) — matrix operand order repaired and checked by reading, not compiled, not run on hardware
//
#include <hip/hip_runtime.h>

typedef __attribute__((ext_vector_type(16))) __bf16 v16bf;
typedef __attribute__((ext_vector_type(8)))  float  v8f;

#define S_LEN   2048
#define NHIST   1024
#define D_LOC   512
#define D_TIM   64
#define D_UID   64
#define D_IN    576      // D_LOC + D_TIM
#define HID     512
#define TGT     1024
#define NLOC    50000
#define KFIN    1088     // 2*HID + D_UID

#define LDSP    40       // padded bf16 row stride in GEMM LDS tiles (32 + 8)
#define TILEB   (128 * LDSP)   // bf16 elements per LDS tile buffer

__device__ __forceinline__ float sigmoidf_(float x) { return 1.0f / (1.0f + expf(-x)); }

// ---------------------------------------------------------------------------
// Generic GEMM: C(MxN,f32) = act( A(MxK,bf16) @ W(NxK,f32)^T + bias )
// A row-major stride K; W row-major stride K (fp32 -> bf16 in LDS).
// Optionally also writes Ct[n*M+m] (transposed f32 copy).
// Block = 256 threads (8 waves, 4x2), block tile 128x128, wave tile 32x64.
// M must be a multiple of 128, K a multiple of 32; N is bounds-checked.
//
// Data movement:
//  - A tile staged with GLOBAL_LOAD_ASYNC_TO_LDS_B128 (ASYNCcnt-tracked DMA,
//    no VGPR round trip); drained with s_wait_asynccnt before the barrier.
//  - W tile prefetched into VGPRs, converted fp32->bf16, stored after WMMAs
//    (out-of-range rows keep a zeroed register tile; store is unconditional
//    so the hot loop has no per-element selects).
//  - Double-buffered LDS: one barrier per K-step, loads overlap WMMA.
// blockIdx.x -> M tiles (fastest) so the 8 blocks sharing one fc_final_w tile
// are adjacent in schedule: 217MB of W streams from HBM ~once, reused via L2.
// ---------------------------------------------------------------------------
__global__ void __launch_bounds__(256) gemm_bf16(
    const __bf16* __restrict__ A, const float* __restrict__ W,
    const float* __restrict__ bias, float* __restrict__ C,
    float* __restrict__ Ct, int M, int N, int K, int act)
{
  __shared__ __bf16 As[2 * TILEB];
  __shared__ __bf16 Ws[2 * TILEB];

  const int m0   = blockIdx.x * 128;
  const int n0   = blockIdx.y * 128;
  const int tid  = threadIdx.x;
  const int lane = tid & 31;
  const int wave = tid >> 5;
  const int waveM = wave & 3;   // 0..3  -> 32 rows each
  const int waveN = wave >> 2;  // 0..1  -> 64 cols each
  const int lmod  = lane & 15;
  const int khalf = lane >> 4;

  v8f acc[2][4];
#pragma unroll
  for (int i = 0; i < 2; ++i)
#pragma unroll
    for (int j = 0; j < 4; ++j) acc[i][j] = (v8f)(0.0f);

  const int rowL  = tid >> 1;   // 0..127 tile row loaded by this thread
  const int halfL = tid & 1;    // which 16-element half of the 32-wide K chunk
  const int ldOff = rowL * LDSP + halfL * 16;

  const __bf16* aSrc = A + (size_t)(m0 + rowL) * K + halfL * 16;
  const bool wValid  = (n0 + rowL) < N;
  const float* wSrc  = W + (size_t)(wValid ? (n0 + rowL) : 0) * K + halfL * 16;

  // 32-bit LDS offsets for the async DMA destination (flat addr bits [31:0])
  const unsigned aLds[2] = {(unsigned)(size_t)(&As[ldOff]),
                            (unsigned)(size_t)(&As[TILEB + ldOff])};

  // wreg stays all-zero for out-of-range rows; stores are unconditional.
  float wreg[16];
#pragma unroll
  for (int e = 0; e < 16; ++e) wreg[e] = 0.0f;

  // ---- prologue: stage K-chunk 0 into buffer 0 ----
  {
    const unsigned long long ga = (unsigned long long)(size_t)aSrc;
    asm volatile("global_load_async_to_lds_b128 %0, %1, off\n\t"
                 "global_load_async_to_lds_b128 %0, %1, off offset:16"
                 :: "v"(aLds[0]), "v"(ga) : "memory");
    if (wValid) {
#pragma unroll
      for (int e = 0; e < 16; ++e) wreg[e] = wSrc[e];
    }
#pragma unroll
    for (int e = 0; e < 16; ++e) Ws[ldOff + e] = (__bf16)wreg[e];
  }
  asm volatile("s_wait_asynccnt 0x0" ::: "memory");
  __syncthreads();

  int buf = 0;
  for (int kc = 0; kc < K; kc += 32) {
    const int nbuf = buf ^ 1;
    const bool hasNext = (kc + 32) < K;

    // issue next tile's loads before computing on the current one
    if (hasNext) {
      const unsigned long long ga =
          (unsigned long long)(size_t)(aSrc + kc + 32);
      asm volatile("global_load_async_to_lds_b128 %0, %1, off\n\t"
                   "global_load_async_to_lds_b128 %0, %1, off offset:16"
                   :: "v"(aLds[nbuf]), "v"(ga) : "memory");
      if (wValid) {
        const float* s = wSrc + kc + 32;
#pragma unroll
        for (int e = 0; e < 16; ++e) wreg[e] = s[e];
      }
    }

    // Fragments per ISA 7.12.2 (16-bit A 16x32; B 32x16 from rows of W).
    const __bf16* aBuf = &As[buf * TILEB];
    const __bf16* wBuf = &Ws[buf * TILEB];
    v16bf afr[2], wfr[4];
#pragma unroll
    for (int ti = 0; ti < 2; ++ti) {
      const int m = waveM * 32 + ti * 16 + lmod;
      const __bf16* p0 = aBuf + m * LDSP + khalf * 8;       // K = khalf*8..
      const __bf16* p1 = aBuf + m * LDSP + 16 + khalf * 8;  // K = 16+khalf*8..
#pragma unroll
      for (int e = 0; e < 8; ++e) { afr[ti][e] = p0[e]; afr[ti][8 + e] = p1[e]; }
    }
#pragma unroll
    for (int tj = 0; tj < 4; ++tj) {
      const int n = waveN * 64 + tj * 16 + lmod;
      const __bf16* p0 = wBuf + n * LDSP + khalf * 8;       // K = khalf*8..
      const __bf16* p1 = wBuf + n * LDSP + 16 + khalf * 8;  // K = 16+khalf*8..
#pragma unroll
      for (int e = 0; e < 8; ++e) { wfr[tj][e] = p0[e]; wfr[tj][8 + e] = p1[e]; }
    }

#pragma unroll
    for (int ti = 0; ti < 2; ++ti)
#pragma unroll
      for (int tj = 0; tj < 4; ++tj)
        acc[ti][tj] = __builtin_amdgcn_wmma_f32_16x16x32_bf16(
            false, afr[ti], false, wfr[tj], (short)0, acc[ti][tj], false, false);

    // convert + store next W tile after WMMAs (hides W load latency);
    // addrspace(3)-clean destination => ds_store_b128, not flat_store.
    if (hasNext) {
      __bf16* dst = &Ws[0] + nbuf * TILEB + ldOff;
#pragma unroll
      for (int e = 0; e < 16; ++e) dst[e] = (__bf16)wreg[e];
    }
    asm volatile("s_wait_asynccnt 0x0" ::: "memory");
    __syncthreads();
    buf = nbuf;
  }

  // Epilogue. C/D layout: VGPR r -> m = r + 8*(lane>=16); n = lane%16.
#pragma unroll
  for (int ti = 0; ti < 2; ++ti) {
#pragma unroll
    for (int tj = 0; tj < 4; ++tj) {
      const int n = n0 + waveN * 64 + tj * 16 + lmod;
      if (n >= N) continue;
      const float b = bias ? bias[n] : 0.0f;
#pragma unroll
      for (int r = 0; r < 8; ++r) {
        const int m = m0 + waveM * 32 + ti * 16 + khalf * 8 + r;
        float v = acc[ti][tj][r] + b;
        if (act == 1) v = tanhf(v);
        C[(size_t)m * N + n] = v;
        if (Ct) Ct[(size_t)n * M + m] = v;
      }
    }
  }
}

// ---------------------------------------------------------------------------
// Prep kernels
// ---------------------------------------------------------------------------
__global__ void seg_offsets(const int* __restrict__ counts, int* __restrict__ offs) {
  const int s = blockIdx.x * blockDim.x + threadIdx.x;
  if (s >= NHIST) return;
  int acc = 0;
  for (int j = 0; j < s; ++j) acc += counts[j];
  offs[s] = acc;
}

__global__ void build_x(const int* __restrict__ loc, const int* __restrict__ tim,
                        const float* __restrict__ emb_loc, const float* __restrict__ emb_tim,
                        __bf16* __restrict__ x) {
  const int idx = blockIdx.x * blockDim.x + threadIdx.x;
  if (idx >= S_LEN * D_IN) return;
  const int i = idx / D_IN, j = idx % D_IN;
  float v = (j < D_LOC) ? emb_loc[(size_t)loc[i] * D_LOC + j]
                        : emb_tim[(size_t)tim[i] * D_TIM + (j - D_LOC)];
  x[idx] = (__bf16)v;
}

__global__ void __launch_bounds__(128) build_hin(
    const int* __restrict__ hloc, const int* __restrict__ htim,
    const int* __restrict__ counts, const int* __restrict__ offs,
    const float* __restrict__ emb_loc, const float* __restrict__ emb_tim,
    __bf16* __restrict__ hin) {
  const int s = blockIdx.x;
  const int off = offs[s];
  const int c = counts[s];
  const float inv = (c > 0) ? 1.0f / (float)c : 0.0f;
  for (int d = threadIdx.x; d < D_LOC; d += 128) {
    float acc = 0.0f;
    for (int cc = 0; cc < c; ++cc)
      acc += emb_loc[(size_t)hloc[off + cc] * D_LOC + d];
    hin[(size_t)s * D_IN + d] = (__bf16)(acc * inv);
  }
  for (int d = threadIdx.x; d < D_TIM; d += 128)
    hin[(size_t)s * D_IN + D_LOC + d] = (__bf16)emb_tim[(size_t)htim[off] * D_TIM + d];
}

__global__ void init_scan(float* __restrict__ hbuf, unsigned int* __restrict__ ctr) {
  for (int i = threadIdx.x; i < 2 * HID; i += blockDim.x) hbuf[i] = 0.0f;
  if (threadIdx.x == 0) *ctr = 0u;
}

// ---------------------------------------------------------------------------
// GRU scan: 32 cooperating blocks, block b owns h columns [b*16, b*16+16).
// w_hh rows {j, 512+j, 1024+j} kept transposed in LDS (fp32, 96KB).
// Cross-block h exchange through L2 with agent-scope atomics + a
// monotonically increasing release/acquire counter (one sync per step).
// ---------------------------------------------------------------------------
#define SCAN_BLOCKS 32
#define SCAN_JW     16        // h columns per block
#define SCAN_ROWS   48        // 3 gates * SCAN_JW
#define SCAN_LDS_FLOATS (HID * SCAN_ROWS + HID + 96 + SCAN_ROWS)

__global__ void __launch_bounds__(128) gru_scan(
    const float* __restrict__ gi, const float* __restrict__ w_hh,
    const float* __restrict__ b_hh, float* __restrict__ hbuf,
    __bf16* __restrict__ outcat, __bf16* __restrict__ qbf,
    unsigned int* __restrict__ ctr)
{
  extern __shared__ float lds[];
  float* wT  = lds;                      // [HID][SCAN_ROWS]
  float* hsh = wT + HID * SCAN_ROWS;     // [HID]
  float* pgh = hsh + HID;                // [96] partial dots
  float* bl  = pgh + 96;                 // [SCAN_ROWS]

  const int tid = threadIdx.x;
  const int b = blockIdx.x;
  const int jBase = b * SCAN_JW;

  // Load transposed w_hh slice: wT[k*48 + rr] = w_hh[(g*512 + jBase + jj)][k]
  for (int idx = tid; idx < HID * SCAN_ROWS; idx += 128) {
    const int rr = idx % SCAN_ROWS, k = idx / SCAN_ROWS;
    const int g = rr >> 4, j = jBase + (rr & 15);
    wT[idx] = w_hh[(size_t)(g * HID + j) * HID + k];
  }
  if (tid < SCAN_ROWS) {
    const int g = tid >> 4, j = jBase + (tid & 15);
    bl[tid] = b_hh[g * HID + j];
  }
  __syncthreads();

  for (int t = 0; t < S_LEN; ++t) {
    const float* cur = hbuf + (t & 1) * HID;
    float* nxt = hbuf + ((t + 1) & 1) * HID;

    // Pull fresh h from L2 (agent-scope loads bypass stale L0 lines).
    for (int k = tid; k < HID; k += 128)
      hsh[k] = __hip_atomic_load(&cur[k], __ATOMIC_RELAXED, __HIP_MEMORY_SCOPE_AGENT);
    __syncthreads();

    // gh partial dots: 96 threads, 2 per row, 256 k's each.
    if (tid < 96) {
      const int rr = tid >> 1, half = tid & 1;
      const int k0 = half * 256;
      float acc = 0.0f;
#pragma unroll 8
      for (int k = 0; k < 256; ++k)
        acc = fmaf(wT[(size_t)(k0 + k) * SCAN_ROWS + rr], hsh[k0 + k], acc);
      pgh[tid] = acc;
    }
    __syncthreads();

    if (tid < SCAN_JW) {
      const int j = jBase + tid;
      const float* g = gi + (size_t)t * (3 * HID);
      const float ghr = pgh[2 * tid] + pgh[2 * tid + 1] + bl[tid];
      const float ghz = pgh[2 * (16 + tid)] + pgh[2 * (16 + tid) + 1] + bl[16 + tid];
      const float ghn = pgh[2 * (32 + tid)] + pgh[2 * (32 + tid) + 1] + bl[32 + tid];
      const float r = sigmoidf_(g[j] + ghr);
      const float z = sigmoidf_(g[HID + j] + ghz);
      const float n = tanhf(g[2 * HID + j] + r * ghn);
      const float hn = (1.0f - z) * n + z * hsh[j];
      __hip_atomic_store(&nxt[j], hn, __ATOMIC_RELAXED, __HIP_MEMORY_SCOPE_AGENT);
      if (t >= S_LEN - TGT) {
        const int row = t - (S_LEN - TGT);
        outcat[(size_t)row * KFIN + j] = (__bf16)hn;   // cols 0..511 of concat
        qbf[(size_t)row * HID + j] = (__bf16)hn;       // dense q for attention
      }
      __threadfence();
    }
    __syncthreads();

    if (tid == 0) {
      __hip_atomic_fetch_add(ctr, 1u, __ATOMIC_RELEASE, __HIP_MEMORY_SCOPE_AGENT);
      const unsigned int target = (unsigned)SCAN_BLOCKS * (unsigned)(t + 1);
      while (__hip_atomic_load(ctr, __ATOMIC_ACQUIRE, __HIP_MEMORY_SCOPE_AGENT) < target)
        __builtin_amdgcn_s_sleep(1);
    }
    __syncthreads();
  }
}

// ---------------------------------------------------------------------------
// Softmax over 1024-wide score rows -> bf16 attention weights
// ---------------------------------------------------------------------------
__global__ void __launch_bounds__(256) softmax_rows(
    const float* __restrict__ S, __bf16* __restrict__ Abf, int ncols) {
  __shared__ float red[256];
  const int row = blockIdx.x;
  const float* x = S + (size_t)row * ncols;
  float mx = -3.4e38f;
  for (int j = threadIdx.x; j < ncols; j += 256) mx = fmaxf(mx, x[j]);
  red[threadIdx.x] = mx; __syncthreads();
  for (int s = 128; s > 0; s >>= 1) {
    if (threadIdx.x < s) red[threadIdx.x] = fmaxf(red[threadIdx.x], red[threadIdx.x + s]);
    __syncthreads();
  }
  mx = red[0]; __syncthreads();
  float sum = 0.0f;
  for (int j = threadIdx.x; j < ncols; j += 256) sum += expf(x[j] - mx);
  red[threadIdx.x] = sum; __syncthreads();
  for (int s = 128; s > 0; s >>= 1) {
    if (threadIdx.x < s) red[threadIdx.x] += red[threadIdx.x + s];
    __syncthreads();
  }
  const float inv = 1.0f / red[0];
  for (int j = threadIdx.x; j < ncols; j += 256)
    Abf[(size_t)row * ncols + j] = (__bf16)(expf(x[j] - mx) * inv);
}

__global__ void fill_outcat(const float* __restrict__ context,
                            const float* __restrict__ emb_uid,
                            const int* __restrict__ uid,
                            __bf16* __restrict__ outcat) {
  const int idx = blockIdx.x * blockDim.x + threadIdx.x;
  if (idx >= TGT * (HID + D_UID)) return;
  const int i = idx / (HID + D_UID), j = idx % (HID + D_UID);
  if (j < HID)
    outcat[(size_t)i * KFIN + HID + j] = (__bf16)context[(size_t)i * HID + j];
  else
    outcat[(size_t)i * KFIN + 2 * HID + (j - HID)] =
        (__bf16)emb_uid[(size_t)uid[0] * D_UID + (j - HID)];
}

__global__ void __launch_bounds__(512) logsoftmax_rows(float* __restrict__ y, int ncols) {
  __shared__ float red[512];
  const int row = blockIdx.x;
  float* x = y + (size_t)row * ncols;
  float mx = -3.4e38f;
  for (int j = threadIdx.x; j < ncols; j += 512) mx = fmaxf(mx, x[j]);
  red[threadIdx.x] = mx; __syncthreads();
  for (int s = 256; s > 0; s >>= 1) {
    if (threadIdx.x < s) red[threadIdx.x] = fmaxf(red[threadIdx.x], red[threadIdx.x + s]);
    __syncthreads();
  }
  mx = red[0]; __syncthreads();
  float sum = 0.0f;
  for (int j = threadIdx.x; j < ncols; j += 512) sum += expf(x[j] - mx);
  red[threadIdx.x] = sum; __syncthreads();
  for (int s = 256; s > 0; s >>= 1) {
    if (threadIdx.x < s) red[threadIdx.x] += red[threadIdx.x + s];
    __syncthreads();
  }
  const float lse = mx + logf(red[0]);
  for (int j = threadIdx.x; j < ncols; j += 512) x[j] = x[j] - lse;
}

// ---------------------------------------------------------------------------
// Host launcher
// ---------------------------------------------------------------------------
static inline size_t alignup_(size_t x) { return (x + 255) & ~(size_t)255; }

extern "C" void kernel_launch(void* const* d_in, const int* in_sizes, int n_in,
                              void* d_out, int out_size, void* d_ws, size_t ws_size,
                              hipStream_t stream) {
  (void)in_sizes; (void)n_in; (void)out_size; (void)ws_size;
  const int* loc        = (const int*)d_in[0];
  const int* tim        = (const int*)d_in[1];
  const int* hloc       = (const int*)d_in[2];
  const int* htim       = (const int*)d_in[3];
  const int* hcount     = (const int*)d_in[4];
  const int* uid        = (const int*)d_in[5];
  /* d_in[6] target_len == TGT (compile-time) */
  const float* emb_loc  = (const float*)d_in[7];
  const float* emb_tim  = (const float*)d_in[8];
  const float* emb_uid  = (const float*)d_in[9];
  const float* fc_attn_w = (const float*)d_in[10];
  const float* fc_attn_b = (const float*)d_in[11];
  const float* w_ih     = (const float*)d_in[12];
  const float* w_hh     = (const float*)d_in[13];
  const float* b_ih     = (const float*)d_in[14];
  const float* b_hh     = (const float*)d_in[15];
  const float* fc_final_w = (const float*)d_in[16];
  const float* fc_final_b = (const float*)d_in[17];
  float* out = (float*)d_out;

  // Workspace carve-up (~31 MB total)
  char* w = (char*)d_ws; size_t off = 0;
  auto take = [&](size_t bytes) { char* p = w + off; off = alignup_(off + bytes); return p; };
  __bf16* xbf     = (__bf16*)take((size_t)S_LEN * D_IN * 2);
  __bf16* hinbf   = (__bf16*)take((size_t)NHIST * D_IN * 2);
  float*  history = (float*) take((size_t)NHIST * HID * 4);
  float*  historyT= (float*) take((size_t)HID * NHIST * 4);
  float*  gi      = (float*) take((size_t)S_LEN * 3 * HID * 4);
  float*  scores  = (float*) take((size_t)TGT * NHIST * 4);
  __bf16* attnbf  = (__bf16*)take((size_t)TGT * NHIST * 2);
  float*  context = (float*) take((size_t)TGT * HID * 4);
  __bf16* outcat  = (__bf16*)take((size_t)TGT * KFIN * 2);
  __bf16* qbf     = (__bf16*)take((size_t)TGT * HID * 2);
  float*  hbuf    = (float*) take((size_t)2 * HID * 4);
  int*    offs    = (int*)   take((size_t)NHIST * 4);
  unsigned int* ctr = (unsigned int*)take(256);

  // 1) segment offsets + input gathers/conversions
  seg_offsets<<<(NHIST + 127) / 128, 128, 0, stream>>>(hcount, offs);
  build_x<<<(S_LEN * D_IN + 255) / 256, 256, 0, stream>>>(loc, tim, emb_loc, emb_tim, xbf);
  build_hin<<<NHIST, 128, 0, stream>>>(hloc, htim, hcount, offs, emb_loc, emb_tim, hinbf);

  // 2) history = tanh(hin @ fc_attn_w^T + b)   (also transposed f32 copy)
  gemm_bf16<<<dim3(NHIST / 128, HID / 128), 256, 0, stream>>>(
      hinbf, fc_attn_w, fc_attn_b, history, historyT, NHIST, HID, D_IN, 1);

  // 3) gi = x @ w_ih^T + b_ih
  gemm_bf16<<<dim3(S_LEN / 128, (3 * HID) / 128), 256, 0, stream>>>(
      xbf, w_ih, b_ih, gi, nullptr, S_LEN, 3 * HID, D_IN, 0);

  // 4) GRU scan (32 cooperating blocks, ~99KB LDS each, fp32 recurrence)
  init_scan<<<1, 256, 0, stream>>>(hbuf, ctr);
  const int scan_lds = SCAN_LDS_FLOATS * 4;
  (void)hipFuncSetAttribute((const void*)gru_scan,
                            hipFuncAttributeMaxDynamicSharedMemorySize, scan_lds);
  gru_scan<<<SCAN_BLOCKS, 128, scan_lds, stream>>>(gi, w_hh, b_hh, hbuf, outcat, qbf, ctr);

  // 5) scores = q @ history^T ; softmax -> bf16 attn
  gemm_bf16<<<dim3(TGT / 128, NHIST / 128), 256, 0, stream>>>(
      qbf, history, nullptr, scores, nullptr, TGT, NHIST, HID, 0);
  softmax_rows<<<TGT, 256, 0, stream>>>(scores, attnbf, NHIST);

  // 6) context = attn @ history  (W = historyT, so W^T = history)
  gemm_bf16<<<dim3(TGT / 128, HID / 128), 256, 0, stream>>>(
      attnbf, historyT, nullptr, context, nullptr, TGT, HID, NHIST, 0);

  // 7) outcat = [q | context | uid_emb]  (q already written by scan)
  fill_outcat<<<(TGT * (HID + D_UID) + 255) / 256, 256, 0, stream>>>(
      context, emb_uid, uid, outcat);

  // 8) y = outcat @ fc_final_w^T + b   (dominant GEMM, 111 GFLOP, bf16 WMMA)
  gemm_bf16<<<dim3(TGT / 128, (NLOC + 127) / 128), 256, 0, stream>>>(
      outcat, fc_final_w, fc_final_b, out, nullptr, TGT, NLOC, KFIN, 0);

  // 9) log_softmax in place
  logsoftmax_rows<<<TGT, 512, 0, stream>>>(out, NLOC);
}
